// GptOssOptExperts_60060822667272
// MI455X (gfx1250) — matrix operand order repaired and checked by reading, not compiled, not run on hardware
//
#include <hip/hip_runtime.h>
#include <hip/hip_bf16.h>

// ---------------------------------------------------------------------------
// GPT-OSS top-2 MoE experts for MI455X (gfx1250), bf16 WMMA pipeline.
//   E=8 experts, H=2048, I=2048, top-K=2, T=2048 tokens, all inputs f32.
// ~103 GFLOP of routed work vs ~420MB of f32 weights. bf16 WMMA
// (v_wmma_f32_16x16x32_bf16, f32 accumulate) is the roofline sweet spot:
// compute ~41us vs HBM floor ~18us at 23.3 TB/s. Weights convert f32->bf16
// while staging through LDS laid out exactly in the ISA WMMA fragment order,
// so fragments load as contiguous ds_load_b128.
// ---------------------------------------------------------------------------

typedef __attribute__((ext_vector_type(16))) __bf16        v16bf;
typedef __attribute__((ext_vector_type(8)))  float         v8f;
typedef __attribute__((ext_vector_type(8)))  unsigned int  v8u;

namespace {
constexpr int   E     = 8;
constexpr int   H     = 2048;
constexpr int   I     = 2048;
constexpr int   TOPK  = 2;
constexpr int   T     = 2048;
constexpr int   TWOI  = 2 * I;
constexpr int   NSLOT = T * TOPK;          // 4096 routed (token, expert) slots
constexpr float ALPHA = 1.702f;
constexpr float LIMIT = 7.0f;

// GEMM tiling: 128 rows x 128 cols x 32 K per block; 8 waves as 4(row)x2(col);
// wave tile = 32x64 -> 2 A-frags x 4 B-frags = 8 v_wmma per wave per K-step.
constexpr int BM = 128;
constexpr int BN = 128;
constexpr int BK = 32;
constexpr int KP = BK / 2;                 // 16 bf16 k-pairs per K-step
constexpr int MAX_TILES = NSLOT / BM;      // 32 row-tiles worst case per expert

// Workspace layout (bytes). Total ~48.2 MB; every byte read is rewritten
// each call (harness does not re-poison between replays).
constexpr size_t WS_OFFS    = 0;                                   // int[E+1]
constexpr size_t WS_TOK     = 256;                                 // int[NSLOT]
constexpr size_t WS_W       = WS_TOK + sizeof(int)   * NSLOT;      // float[NSLOT]
constexpr size_t WS_SLOTOF  = WS_W   + sizeof(float) * NSLOT;      // int[NSLOT]
constexpr size_t WS_GATED   = 65536;                               // bf16[NSLOT*I]
constexpr size_t WS_CONTRIB = WS_GATED + (size_t)NSLOT * I * 2;    // f32[NSLOT*H]

// f32 -> bf16 (round-half-up) pair pack: one v_add per float + one v_perm.
__device__ __forceinline__ unsigned int pack2(float lo, float hi) {
    const unsigned int ul = __float_as_uint(lo) + 0x8000u;
    const unsigned int uh = __float_as_uint(hi) + 0x8000u;
    return __builtin_amdgcn_perm(uh, ul, 0x07060302u);  // {uh[31:16], ul[31:16]}
}
__device__ __forceinline__ unsigned short f2bf(float f) {
    return (unsigned short)((__float_as_uint(f) + 0x8000u) >> 16);
}

// Stage a 4(k) x 4(col) f32 patch into fragment-major LDS B [col][k_pair].
__device__ __forceinline__ void store_b_patch(unsigned int* lds_b, int kpp, int c0,
                                              const float4 r0, const float4 r1,
                                              const float4 r2, const float4 r3) {
    *(uint2*)&lds_b[(c0 + 0) * KP + 2 * kpp] = make_uint2(pack2(r0.x, r1.x), pack2(r2.x, r3.x));
    *(uint2*)&lds_b[(c0 + 1) * KP + 2 * kpp] = make_uint2(pack2(r0.y, r1.y), pack2(r2.y, r3.y));
    *(uint2*)&lds_b[(c0 + 2) * KP + 2 * kpp] = make_uint2(pack2(r0.z, r1.z), pack2(r2.z, r3.z));
    *(uint2*)&lds_b[(c0 + 3) * KP + 2 * kpp] = make_uint2(pack2(r0.w, r1.w), pack2(r2.w, r3.w));
}

// Per-lane fragment fetch: ISA 16-bit A 16x32 layout (lane<16: K0..7,16..23;
// lane>=16: +8). Row stride in LDS = KP uints. Two contiguous b128 loads.
__device__ __forceinline__ v8u load_a_frag(const unsigned int* lds_a, int lr, int lane) {
    const int off = lr * KP + (lane >> 4) * 4;
    const uint4 lo = *(const uint4*)&lds_a[off];
    const uint4 hi = *(const uint4*)&lds_a[off + 8];
    v8u u; u[0] = lo.x; u[1] = lo.y; u[2] = lo.z; u[3] = lo.w;
    u[4] = hi.x; u[5] = hi.y; u[6] = hi.z; u[7] = hi.w;
    return u;
}
// B fragment: lane<16 -> K0..7,16..23, lane>=16 -> +8, col = lane%16.
// Fragment-major LDS makes this two contiguous b128 loads too.
__device__ __forceinline__ v8u load_b_frag(const unsigned int* lds_b, int n, int lane) {
    const int off = n * KP + (lane >> 4) * 4;
    const uint4 lo = *(const uint4*)&lds_b[off];
    const uint4 hi = *(const uint4*)&lds_b[off + 8];
    v8u u; u[0] = lo.x; u[1] = lo.y; u[2] = lo.z; u[3] = lo.w;
    u[4] = hi.x; u[5] = hi.y; u[6] = hi.z; u[7] = hi.w;
    return u;
}
} // namespace

// ---------------------------------------------------------------------------
// Kernel 1: routing. Group the 4096 (token,k) occurrences by expert and
// record the inverse map so the final combine is deterministic (no float
// atomics anywhere in the pipeline).
// ---------------------------------------------------------------------------
__global__ __launch_bounds__(256)
void k_route(const int* __restrict__ ridx, const float* __restrict__ rw,
             int* __restrict__ offs, int* __restrict__ slot_tok,
             float* __restrict__ slot_w, int* __restrict__ slot_of) {
    __shared__ int cnt[E];
    __shared__ int cur[E];
    const int tid = threadIdx.x;
    if (tid < E) cnt[tid] = 0;
    __syncthreads();
    for (int n = tid; n < NSLOT; n += 256) atomicAdd(&cnt[ridx[n]], 1);
    __syncthreads();
    if (tid == 0) {
        int run = 0;
        for (int e = 0; e < E; ++e) { offs[e] = run; cur[e] = run; run += cnt[e]; }
        offs[E] = run;
    }
    __syncthreads();
    for (int n = tid; n < NSLOT; n += 256) {
        const int e = ridx[n];
        const int p = atomicAdd(&cur[e], 1);
        const int t = n / TOPK;
        slot_tok[p] = t;
        slot_w[p]   = rw[t * E + e];
        slot_of[n]  = p;
    }
}

// ---------------------------------------------------------------------------
// Kernel 2: gate_up GEMM + clamp/SiLU-GLU -> gated activations (bf16).
// Software-pipelined: next tile's global_load_b128s issue right after the
// store barrier and fly during the 64-WMMA burst of the current K-step.
// ---------------------------------------------------------------------------
__global__ __launch_bounds__(256)
void k_gateup(const float* __restrict__ x, const float* __restrict__ gup,
              const float* __restrict__ gup_b, const int* __restrict__ offs,
              const int* __restrict__ slot_tok, unsigned short* __restrict__ gated) {
    __shared__ unsigned int lds_a[BM * KP];   // 128 rows x 16 k-pairs (8KB)
    __shared__ unsigned int lds_b[BN * KP];   // 128 cols x 16 k-pairs (8KB)

    const int e  = blockIdx.x / MAX_TILES;
    const int tr = blockIdx.x % MAX_TILES;
    const int base = offs[e];
    const int rows = offs[e + 1] - base;
    if (tr * BM >= rows) return;                       // uniform early-exit
    int rows_rem = rows - tr * BM; if (rows_rem > BM) rows_rem = BM;
    const int slot_base = base + tr * BM;
    const int n0 = blockIdx.y * BN;

    const int tid  = threadIdx.x;
    const int lane = tid & 31;
    const int wave = tid >> 5;
    const int wm   = wave & 3;                          // 4 row-waves
    const int wn   = wave >> 2;                         // 2 col-waves

    // A loader: row = tid/2 (0..127), half = tid&1 -> 16 K values.
    const int arow = tid >> 1, ahalf = tid & 1;
    const int tokA = (arow < rows_rem) ? slot_tok[slot_base + arow] : -1;
    const float* xrow = x + (size_t)(tokA < 0 ? 0 : tokA) * H + ahalf * 16;
    // B loader: kpp = tid/32 (4 K rows), c0 = (tid&31)*4 (4 cols).
    const int kpp = tid >> 5, c0 = (tid & 31) * 4;
    const float* gb = gup + (size_t)e * H * TWOI + (size_t)(4 * kpp) * TWOI + n0 + c0;

    v8f zero = {};
    v8f acc[2][4];
#pragma unroll
    for (int m = 0; m < 2; ++m)
#pragma unroll
        for (int n = 0; n < 4; ++n) acc[m][n] = zero;

    const int NKS = H / BK;
    float4 ag0, ag1, ag2, ag3;            // staged A (16 f32)
    float4 bg0, bg1, bg2, bg3;            // staged B (4 K-rows x 4 cols)
    // prologue loads (K-step 0)
    if (tokA >= 0) {
        ag0 = *(const float4*)(xrow + 0); ag1 = *(const float4*)(xrow + 4);
        ag2 = *(const float4*)(xrow + 8); ag3 = *(const float4*)(xrow + 12);
    } else { ag0 = ag1 = ag2 = ag3 = make_float4(0.f, 0.f, 0.f, 0.f); }
    bg0 = *(const float4*)(gb);
    bg1 = *(const float4*)(gb + TWOI);
    bg2 = *(const float4*)(gb + 2 * TWOI);
    bg3 = *(const float4*)(gb + 3 * TWOI);

    for (int kt = 0; kt < NKS; ++kt) {
        { // stage A: packed bf16 pairs, row-major [row][k_pair]
            unsigned int* dst = &lds_a[arow * KP + ahalf * 8];
            uint4 w0, w1;
            w0.x = pack2(ag0.x, ag0.y); w0.y = pack2(ag0.z, ag0.w);
            w0.z = pack2(ag1.x, ag1.y); w0.w = pack2(ag1.z, ag1.w);
            w1.x = pack2(ag2.x, ag2.y); w1.y = pack2(ag2.z, ag2.w);
            w1.z = pack2(ag3.x, ag3.y); w1.w = pack2(ag3.z, ag3.w);
            *(uint4*)dst = w0; *(uint4*)(dst + 4) = w1;
        }
        store_b_patch(lds_b, kpp, c0, bg0, bg1, bg2, bg3);
        __syncthreads();

        // issue next K-step's global loads (latency hidden under WMMAs)
        if (kt + 1 < NKS) {
            const int kh = (kt + 1) * BK;
            if (tokA >= 0) {
                ag0 = *(const float4*)(xrow + kh + 0); ag1 = *(const float4*)(xrow + kh + 4);
                ag2 = *(const float4*)(xrow + kh + 8); ag3 = *(const float4*)(xrow + kh + 12);
            }
            const float* b = gb + (size_t)kh * TWOI;
            bg0 = *(const float4*)(b);
            bg1 = *(const float4*)(b + TWOI);
            bg2 = *(const float4*)(b + 2 * TWOI);
            bg3 = *(const float4*)(b + 3 * TWOI);
            if (kt + 2 < NKS)   // gfx1250 global_prefetch on the weight stream
                __builtin_prefetch(gb + (size_t)(kt + 2) * BK * TWOI, 0, 3);
        }

        v16bf af[2];
#pragma unroll
        for (int m = 0; m < 2; ++m)
            af[m] = __builtin_bit_cast(v16bf, load_a_frag(lds_a, wm * 32 + m * 16 + (lane & 15), lane));
#pragma unroll
        for (int n = 0; n < 4; ++n) {
            const v16bf bf = __builtin_bit_cast(v16bf, load_b_frag(lds_b, wn * 64 + n * 16 + (lane & 15), lane));
#pragma unroll
            for (int m = 0; m < 2; ++m)
                acc[m][n] = __builtin_amdgcn_wmma_f32_16x16x32_bf16(
                    false, af[m], false, bf, (short)0, acc[m][n], false, false);
        }
        __syncthreads();
    }

    // Epilogue: + bias, pair gate(even col)/up(odd col) via lane shuffle,
    // clamp + sigmoid-GLU, bf16 store of gated activations.
    const float* be = gup_b + e * TWOI;
#pragma unroll
    for (int m = 0; m < 2; ++m) {
#pragma unroll
        for (int n = 0; n < 4; ++n) {
#pragma unroll
            for (int r = 0; r < 8; ++r) {
                const int row  = wm * 32 + m * 16 + r + (lane >> 4) * 8;
                const int ncol = wn * 64 + n * 16 + (lane & 15);
                const int g    = n0 + ncol;
                float v = acc[m][n][r] + be[g];
                const float p = __shfl_xor(v, 1, 32);   // partner column
                if (((lane & 1) == 0) && row < rows_rem) {
                    const float gate = fminf(v, LIMIT);
                    const float up   = fminf(fmaxf(p, -LIMIT), LIMIT);
                    const float glu  = gate / (1.0f + __expf(-ALPHA * gate));
                    gated[(size_t)(slot_base + row) * I + (g >> 1)] = f2bf((up + 1.0f) * glu);
                }
            }
        }
    }
}

// ---------------------------------------------------------------------------
// Kernel 3: down GEMM. A = gated (already bf16 -> pure LDS copy),
// B = down_proj[e] (f32->bf16). Writes w*(acc+bias) per slot to contrib.
// ---------------------------------------------------------------------------
__global__ __launch_bounds__(256)
void k_down(const unsigned short* __restrict__ gated, const float* __restrict__ dp,
            const float* __restrict__ dp_b, const int* __restrict__ offs,
            const float* __restrict__ slot_w, float* __restrict__ contrib) {
    __shared__ unsigned int lds_a[BM * KP];
    __shared__ unsigned int lds_b[BN * KP];

    const int e  = blockIdx.x / MAX_TILES;
    const int tr = blockIdx.x % MAX_TILES;
    const int base = offs[e];
    const int rows = offs[e + 1] - base;
    if (tr * BM >= rows) return;
    int rows_rem = rows - tr * BM; if (rows_rem > BM) rows_rem = BM;
    const int slot_base = base + tr * BM;
    const int h0 = blockIdx.y * BN;

    const int tid  = threadIdx.x;
    const int lane = tid & 31;
    const int wave = tid >> 5;
    const int wm   = wave & 3;
    const int wn   = wave >> 2;

    const int arow = tid >> 1, ahalf = tid & 1;
    const unsigned int* grow = (const unsigned int*)gated
        + ((size_t)(slot_base + arow) * I) / 2 + ahalf * 8;
    const bool avalid = (arow < rows_rem);
    const int kpp = tid >> 5, c0 = (tid & 31) * 4;
    const float* db = dp + (size_t)e * I * H + (size_t)(4 * kpp) * H + h0 + c0;

    v8f zero = {};
    v8f acc[2][4];
#pragma unroll
    for (int m = 0; m < 2; ++m)
#pragma unroll
        for (int n = 0; n < 4; ++n) acc[m][n] = zero;

    const int NKS = I / BK;
    uint4 ag0, ag1;
    float4 bg0, bg1, bg2, bg3;
    if (avalid) { ag0 = *(const uint4*)(grow); ag1 = *(const uint4*)(grow + 4); }
    else        { ag0 = ag1 = make_uint4(0u, 0u, 0u, 0u); }
    bg0 = *(const float4*)(db);
    bg1 = *(const float4*)(db + H);
    bg2 = *(const float4*)(db + 2 * H);
    bg3 = *(const float4*)(db + 3 * H);

    for (int kt = 0; kt < NKS; ++kt) {
        {
            unsigned int* dst = &lds_a[arow * KP + ahalf * 8];
            *(uint4*)dst = ag0; *(uint4*)(dst + 4) = ag1;
        }
        store_b_patch(lds_b, kpp, c0, bg0, bg1, bg2, bg3);
        __syncthreads();

        if (kt + 1 < NKS) {
            const int kh = (kt + 1) * BK;
            if (avalid) {
                ag0 = *(const uint4*)(grow + kh / 2);
                ag1 = *(const uint4*)(grow + kh / 2 + 4);
            }
            const float* b = db + (size_t)kh * H;
            bg0 = *(const float4*)(b);
            bg1 = *(const float4*)(b + H);
            bg2 = *(const float4*)(b + 2 * H);
            bg3 = *(const float4*)(b + 3 * H);
            if (kt + 2 < NKS)
                __builtin_prefetch(db + (size_t)(kt + 2) * BK * H, 0, 3);
        }

        v16bf af[2];
#pragma unroll
        for (int m = 0; m < 2; ++m)
            af[m] = __builtin_bit_cast(v16bf, load_a_frag(lds_a, wm * 32 + m * 16 + (lane & 15), lane));
#pragma unroll
        for (int n = 0; n < 4; ++n) {
            const v16bf bf = __builtin_bit_cast(v16bf, load_b_frag(lds_b, wn * 64 + n * 16 + (lane & 15), lane));
#pragma unroll
            for (int m = 0; m < 2; ++m)
                acc[m][n] = __builtin_amdgcn_wmma_f32_16x16x32_bf16(
                    false, af[m], false, bf, (short)0, acc[m][n], false, false);
        }
        __syncthreads();
    }

    const float* bb = dp_b + e * H;
#pragma unroll
    for (int m = 0; m < 2; ++m) {
#pragma unroll
        for (int n = 0; n < 4; ++n) {
#pragma unroll
            for (int r = 0; r < 8; ++r) {
                const int row  = wm * 32 + m * 16 + r + (lane >> 4) * 8;
                const int ncol = wn * 64 + n * 16 + (lane & 15);
                const int h    = h0 + ncol;
                if (row < rows_rem) {
                    const int slot = slot_base + row;
                    contrib[(size_t)slot * H + h] = (acc[m][n][r] + bb[h]) * slot_w[slot];
                }
            }
        }
    }
}

// ---------------------------------------------------------------------------
// Kernel 4: deterministic combine: out[t] = contrib[slot(t,0)] + contrib[slot(t,1)].
// Writes every output element -> no zero-init, no atomics.
// ---------------------------------------------------------------------------
__global__ __launch_bounds__(256)
void k_combine(const float* __restrict__ contrib, const int* __restrict__ slot_of,
               float* __restrict__ out) {
    const int t  = blockIdx.x;
    const int s0 = slot_of[t * TOPK + 0];
    const int s1 = slot_of[t * TOPK + 1];
    const float4* c0 = (const float4*)(contrib + (size_t)s0 * H);
    const float4* c1 = (const float4*)(contrib + (size_t)s1 * H);
    float4* o = (float4*)(out + (size_t)t * H);
    for (int i = threadIdx.x; i < H / 4; i += 256) {
        const float4 a = c0[i];
        const float4 b = c1[i];
        o[i] = make_float4(a.x + b.x, a.y + b.y, a.z + b.z, a.w + b.w);
    }
}

// ---------------------------------------------------------------------------
extern "C" void kernel_launch(void* const* d_in, const int* in_sizes, int n_in,
                              void* d_out, int out_size, void* d_ws, size_t ws_size,
                              hipStream_t stream) {
    const float* x    = (const float*)d_in[0];   // hidden_states (1,2048,2048)
    const int*   ridx = (const int*)  d_in[1];   // router_indices (4096)
    const float* rw   = (const float*)d_in[2];   // routing_weights (2048,8)
    const float* gup  = (const float*)d_in[3];   // gate_up_proj (8,2048,4096)
    const float* gupb = (const float*)d_in[4];   // gate_up_proj_bias (8,4096)
    const float* dp   = (const float*)d_in[5];   // down_proj (8,2048,2048)
    const float* dpb  = (const float*)d_in[6];   // down_proj_bias (8,2048)
    float* out = (float*)d_out;

    char* ws = (char*)d_ws;
    int*            offs    = (int*)(ws + WS_OFFS);
    int*            stok    = (int*)(ws + WS_TOK);
    float*          sw      = (float*)(ws + WS_W);
    int*            sof     = (int*)(ws + WS_SLOTOF);
    unsigned short* gated   = (unsigned short*)(ws + WS_GATED);
    float*          contrib = (float*)(ws + WS_CONTRIB);

    k_route<<<1, 256, 0, stream>>>(ridx, rw, offs, stok, sw, sof);
    k_gateup<<<dim3(E * MAX_TILES, TWOI / BN), 256, 0, stream>>>(
        x, gup, gupb, offs, stok, gated);
    k_down<<<dim3(E * MAX_TILES, H / BN), 256, 0, stream>>>(
        gated, dp, dpb, offs, sw, contrib);
    k_combine<<<T, 256, 0, stream>>>(contrib, sof, out);
}
